// HMM_61830349193862
// MI455X (gfx1250) — matrix operand order repaired and checked by reading, not compiled, not run on hardware
//
#include <hip/hip_runtime.h>

typedef __attribute__((ext_vector_type(16))) _Float16 v16h;
typedef __attribute__((ext_vector_type(8)))  float    v8f;

#define S    1024
#define NOBS 4096
#define TT   8192
#define NB   64      // persistent blocks (all resident -> grid barrier is safe)
#define BS   256     // 8 wave32 per block
#define SPB  16      // output states per block
#define NW   8       // waves per block

__device__ __forceinline__ void grid_barrier(unsigned* cnt, unsigned* gen) {
  __syncthreads();
  if (threadIdx.x == 0) {
    unsigned g = __hip_atomic_load(gen, __ATOMIC_RELAXED, __HIP_MEMORY_SCOPE_AGENT);
    unsigned a = __hip_atomic_fetch_add(cnt, 1u, __ATOMIC_ACQ_REL, __HIP_MEMORY_SCOPE_AGENT);
    if (a == NB - 1u) {
      __hip_atomic_store(cnt, 0u, __ATOMIC_RELAXED, __HIP_MEMORY_SCOPE_AGENT);
      __hip_atomic_fetch_add(gen, 1u, __ATOMIC_RELEASE, __HIP_MEMORY_SCOPE_AGENT);
    } else {
      while (__hip_atomic_load(gen, __ATOMIC_ACQUIRE, __HIP_MEMORY_SCOPE_AGENT) == g)
        __builtin_amdgcn_s_sleep(2);
    }
  }
  __syncthreads();
}

// Wave32 max reduction, no barriers (lowers to ds_swizzle/permlane cross-lane ops).
__device__ __forceinline__ float wave_max(float v) {
#pragma unroll
  for (int m = 16; m > 0; m >>= 1) v = fmaxf(v, __shfl_xor(v, m, 32));
  return v;
}
__device__ __forceinline__ float wave_sum(float v) {
#pragma unroll
  for (int m = 16; m > 0; m >>= 1) v += __shfl_xor(v, m, 32);
  return v;
}

// Build exp(trans)^T in f16: expT[n*S + i] = exp(trans[i*S + n]); reset barrier state.
__global__ void hmm_prep(const float* __restrict__ trans, _Float16* __restrict__ expT,
                         unsigned* cnt, unsigned* gen) {
  unsigned idx = blockIdx.x * BS + threadIdx.x;
  if (idx == 0) { *cnt = 0u; *gen = 0u; }
  unsigned n = idx >> 10, i = idx & (S - 1);
  expT[idx] = (_Float16)__expf(trans[i * S + n]);
}

__global__ void __launch_bounds__(BS) hmm_forward(
    const int* __restrict__ obs, const float* __restrict__ start,
    const float* __restrict__ emit, const _Float16* __restrict__ expT,
    float* __restrict__ laBuf, unsigned* cnt, unsigned* gen, float* __restrict__ out) {
  __shared__ __align__(32) _Float16 Atile[SPB][S];   // 32 KB: this block's slab, resident all run
  __shared__ float                  laS[S];          // 4 KB
  __shared__ __align__(32) _Float16 alphaS[S];       // 2 KB
  __shared__ float                  partial[NW][SPB];
  __shared__ float                  red[NW];

  const int tid  = threadIdx.x;
  const int blk  = blockIdx.x;
  const int lane = tid & 31;           // wave32
  const int wv   = tid >> 5;           // 8 waves

  // Stage 16 rows of exp(trans)^T into LDS once (coalesced along i).
  for (int e = tid; e < SPB * S; e += BS) {
    int m = e >> 10, i = e & (S - 1);
    Atile[m][i] = expT[(blk * SPB + m) * S + i];
  }
  __syncthreads();

  // Loop-invariant A fragments: 16x32 f16 tiles, contiguous K order matching B:
  // lanes 0-15 hold K=0..15, lanes 16-31 hold K=16..31.
  const int am  = lane & 15;
  const int abk = (lane < 16) ? 0 : 16;
  v16h regA[4];
  for (int c = 0; c < 4; ++c) {
    int kbase = (wv * 4 + c) * 32;
    for (int j = 0; j < 8; ++j) {
      regA[c][j]     = Atile[am][kbase + abk + j];
      regA[c][8 + j] = Atile[am][kbase + abk + 8 + j];
    }
  }

  // t = 0: la0 = start + emit[:, obs[0]]
  if (tid < SPB) {
    int st = blk * SPB + tid;
    float v = start[st] + emit[st * NOBS + obs[0]];
    __hip_atomic_store(&laBuf[st], v, __ATOMIC_RELAXED, __HIP_MEMORY_SCOPE_AGENT);
  }
  grid_barrier(cnt, gen);

  const int boff = (lane < 16) ? 0 : 16;  // 16-bit B 32x16: lanes 0-15 K=0..15, 16-31 K=16..31

  for (int t = 1; t < TT; ++t) {
    const int p = t & 1;
    const float* src = laBuf + (p ^ 1) * S;

    // Fused: pull la vector into LDS and fold running max in the same pass.
    float mx = -3.4e38f;
#pragma unroll
    for (int i = tid; i < S; i += BS) {
      float v = __hip_atomic_load(&src[i], __ATOMIC_RELAXED, __HIP_MEMORY_SCOPE_AGENT);
      laS[i] = v;
      mx = fmaxf(mx, v);
    }
    mx = wave_max(mx);                       // barrier-free in-wave reduce
    if (lane == 0) red[wv] = mx;
    __syncthreads();
    float gmax = red[0];
#pragma unroll
    for (int w = 1; w < NW; ++w) gmax = fmaxf(gmax, red[w]);

#pragma unroll
    for (int i = tid; i < S; i += BS)
      alphaS[i] = (_Float16)__expf(laS[i] - gmax);
    __syncthreads();

    // matvec via WMMA: A = slab rows (registers), B = alpha chunk broadcast across 16 cols
    v8f acc = {};
#pragma unroll
    for (int c = 0; c < 4; ++c) {
      const int kbase = (wv * 4 + c) * 32;
      v16h b = *(const v16h*)(&alphaS[kbase + boff]);
      acc = __builtin_amdgcn_wmma_f32_16x16x32_f16(false, regA[c], false, b,
                                                   (short)0, acc, false, false);
    }
    // D column 0: lane 0 holds M=0..7, lane 16 holds M=8..15 (single-branch store)
    if ((lane & 15) == 0) {
      float* dst = &partial[wv][(lane >> 4) * 8];
#pragma unroll
      for (int v = 0; v < 8; ++v) dst[v] = acc[v];
    }
    __syncthreads();

    if (tid < SPB) {
      float s = 0.f;
#pragma unroll
      for (int w = 0; w < NW; ++w) s += partial[w][tid];
      int st = blk * SPB + tid;
      float nla = gmax + __logf(s) + emit[st * NOBS + obs[t]];
      __hip_atomic_store(&laBuf[p * S + st], nla, __ATOMIC_RELAXED, __HIP_MEMORY_SCOPE_AGENT);
    }
    grid_barrier(cnt, gen);
  }

  // Final logsumexp over la[T-1] by block 0.
  if (blk == 0) {
    const float* fin = laBuf + ((TT - 1) & 1) * S;
    float mx = -3.4e38f;
#pragma unroll
    for (int i = tid; i < S; i += BS) {
      float v = __hip_atomic_load(&fin[i], __ATOMIC_RELAXED, __HIP_MEMORY_SCOPE_AGENT);
      laS[i] = v;
      mx = fmaxf(mx, v);
    }
    mx = wave_max(mx);
    if (lane == 0) red[wv] = mx;
    __syncthreads();
    float gmax = red[0];
#pragma unroll
    for (int w = 1; w < NW; ++w) gmax = fmaxf(gmax, red[w]);
    __syncthreads();

    float sm = 0.f;
#pragma unroll
    for (int i = tid; i < S; i += BS) sm += __expf(laS[i] - gmax);
    sm = wave_sum(sm);
    if (lane == 0) red[wv] = sm;
    __syncthreads();
    if (tid == 0) {
      float s = 0.f;
#pragma unroll
      for (int w = 0; w < NW; ++w) s += red[w];
      out[0] = gmax + __logf(s);
    }
  }
}

extern "C" void kernel_launch(void* const* d_in, const int* in_sizes, int n_in,
                              void* d_out, int out_size, void* d_ws, size_t ws_size,
                              hipStream_t stream) {
  const int*   obs   = (const int*)d_in[0];
  const float* start = (const float*)d_in[1];
  const float* trans = (const float*)d_in[2];
  const float* emit  = (const float*)d_in[3];

  char* ws = (char*)d_ws;
  _Float16* expT  = (_Float16*)ws;                               // 2 MB: exp(trans)^T f16
  float*    laBuf = (float*)(ws + 2u * 1024u * 1024u);           // 8 KB: double-buffered log-alpha
  unsigned* cnt   = (unsigned*)(ws + 2u * 1024u * 1024u + 8192u);
  unsigned* gen   = cnt + 1;

  hmm_prep<<<(S * S) / BS, BS, 0, stream>>>(trans, expT, cnt, gen);
  hmm_forward<<<NB, BS, 0, stream>>>(obs, start, emit, expT, laBuf, cnt, gen,
                                     (float*)d_out);
}
